// RelationModule_2001454760451
// MI455X (gfx1250) — compile-verified, matrix operand order checked by reading, not run on hardware
//
#include <hip/hip_runtime.h>

#define NN    1024
#define FEAT  2048
#define EMB   64
#define FC_DIM 16
#define GROUP 16
#define DIMG  128   // FEAT / GROUP

typedef _Float16 half_t;
typedef __attribute__((ext_vector_type(16))) _Float16 v16h;
typedef __attribute__((ext_vector_type(8)))  _Float16 v8h;
typedef __attribute__((ext_vector_type(4)))  _Float16 v4h;
typedef __attribute__((ext_vector_type(8)))  float    v8f;
typedef __attribute__((ext_vector_type(4)))  float    v4f;
typedef __attribute__((ext_vector_type(4)))  int      v4i;

// CDNA5 async global->LDS copy (ASYNCcnt-tracked, bypasses VGPRs), with a
// synchronous fallback if the builtin is unavailable on this toolchain.
#if __has_builtin(__builtin_amdgcn_global_load_async_to_lds_b128)
#define USE_ASYNC_LDS 1
typedef __attribute__((address_space(1))) v4i* as1_v4i_ptr;
typedef __attribute__((address_space(3))) v4i* as3_v4i_ptr;
#endif

__device__ __forceinline__ void stage16(half_t* dst_lds, const half_t* src_glb) {
#ifdef USE_ASYNC_LDS
  __builtin_amdgcn_global_load_async_to_lds_b128(
      (as1_v4i_ptr)(src_glb),
      (as3_v4i_ptr)(dst_lds),
      /*imm offset*/ 0, /*cpol*/ 0);
#else
  *(v8h*)dst_lds = *(const v8h*)src_glb;
#endif
}

__device__ __forceinline__ void wait_async_lds() {
#ifdef USE_ASYNC_LDS
#if __has_builtin(__builtin_amdgcn_s_wait_asynccnt)
  __builtin_amdgcn_s_wait_asynccnt(0);
#else
  asm volatile("s_wait_asynccnt 0x0" ::: "memory");
#endif
#endif
}

// Build a 16xK f16 WMMA operand fragment per the CDNA5 ISA layout:
// lane l: row/col = l&15, half h = l>>4; holds K = {h*8..h*8+7, 16+h*8..16+h*8+7}.
// Caller passes p = rowbase + h*8 (16B aligned); chunks at +0 and +16 halfs.
__device__ __forceinline__ v16h load_frag16(const half_t* p) {
  v8h lo = *(const v8h*)(p);
  v8h hi = *(const v8h*)(p + 16);
  v16h f;
#pragma unroll
  for (int i = 0; i < 8; ++i) { f[i] = lo[i]; f[i + 8] = hi[i]; }
  return f;
}

// ---------------------------------------------------------------------------
// Kernel 1: logw[n][g][m] = log(max(pe[n,m,:]·fc_w[g,:] + fc_b[g], 1e-3))
// (relu folded into the clamp). Memory bound: 256MB read, 64MB write.
// ---------------------------------------------------------------------------
__global__ __launch_bounds__(256) void posbias_kernel(
    const float* __restrict__ pe, const float* __restrict__ fc_w,
    const float* __restrict__ fc_b, float* __restrict__ logw) {
  __shared__ float w_s[FC_DIM][EMB];
  __shared__ float b_s[FC_DIM];
  int tid = threadIdx.x;
  for (int i = tid; i < FC_DIM * EMB; i += 256) w_s[i >> 6][i & 63] = fc_w[i];
  if (tid < FC_DIM) b_s[tid] = fc_b[tid];
  __syncthreads();

  long long idx = (long long)blockIdx.x * 256 + tid;  // 0 .. N*NONGT-1
  int n = (int)(idx >> 10);
  int m = (int)(idx & 1023);
  const float* p = pe + idx * EMB;
  v4f x[16];
#pragma unroll
  for (int i = 0; i < 16; ++i) x[i] = *(const v4f*)(p + i * 4);

#pragma unroll
  for (int g = 0; g < FC_DIM; ++g) {
    float acc = b_s[g];
#pragma unroll
    for (int i = 0; i < 16; ++i) {
      const float* wr = &w_s[g][i * 4];
      acc += x[i][0] * wr[0] + x[i][1] * wr[1] + x[i][2] * wr[2] + x[i][3] * wr[3];
    }
    logw[(((long long)n * GROUP + g) << 10) + m] = __logf(fmaxf(acc, 1e-3f));
  }
}

// ---------------------------------------------------------------------------
// Kernel 2: WMMA GEMM  O[n,j] = sum_f A[n,f] * W[j,f] (+ bias[j]), f16 out.
// A: [1024,2048] f32, W: [2048,2048] f32 row-major (B-operand = W rows).
// TRANS_OUT: store Out[j*1024 + n] (for VpT) via an LDS transpose.
// Block tile 128(M) x 64(N), K step 64; 8 waves, each a 32x32 C tile.
// Register-stage double buffering: prefetch next f32 tiles during compute.
// ---------------------------------------------------------------------------
template <bool TRANS_OUT, bool HAS_BIAS>
__global__ __launch_bounds__(256) void gemm_nt_kernel(
    const float* __restrict__ A, const float* __restrict__ W,
    const float* __restrict__ bias, half_t* __restrict__ Out) {
  constexpr int BM = 128, BN = 64, BK = 64;
  constexpr int LDA = BK + 8;  // 72 halfs = 144B: 16B-aligned rows, conflict-free frag reads
  __shared__ half_t As[BM * LDA];
  __shared__ half_t Bs[BN * LDA];

  int tid  = threadIdx.x;
  int wave = tid >> 5, lane = tid & 31;
  int r = lane & 15, h = lane >> 4;
  int m0 = blockIdx.x * BM;
  int n0 = blockIdx.y * BN;
  int wm = (wave & 3) * 32;   // wave M offset within block
  int wn = (wave >> 2) * 32;  // wave N offset within block

  v8f c[2][2] = {};
  v4f ra[8], rb[4];

  auto loadRegs = [&](int kk) {
#pragma unroll
    for (int it = 0; it < 8; ++it) {        // A: 128 rows x 64 f32 = 2048 float4
      int id = tid + it * 256;
      int row = id >> 4, col = (id & 15) * 4;
      ra[it] = *(const v4f*)(A + (long long)(m0 + row) * FEAT + kk + col);
    }
#pragma unroll
    for (int it = 0; it < 4; ++it) {        // B: 64 rows x 64 f32 = 1024 float4
      int id = tid + it * 256;
      int row = id >> 4, col = (id & 15) * 4;
      rb[it] = *(const v4f*)(W + (long long)(n0 + row) * FEAT + kk + col);
    }
  };
  auto storeLDS = [&]() {
#pragma unroll
    for (int it = 0; it < 8; ++it) {
      int id = tid + it * 256;
      int row = id >> 4, col = (id & 15) * 4;
      v4h hv = { (half_t)ra[it][0], (half_t)ra[it][1],
                 (half_t)ra[it][2], (half_t)ra[it][3] };
      *(v4h*)&As[row * LDA + col] = hv;
    }
#pragma unroll
    for (int it = 0; it < 4; ++it) {
      int id = tid + it * 256;
      int row = id >> 4, col = (id & 15) * 4;
      v4h hv = { (half_t)rb[it][0], (half_t)rb[it][1],
                 (half_t)rb[it][2], (half_t)rb[it][3] };
      *(v4h*)&Bs[row * LDA + col] = hv;
    }
  };

  loadRegs(0);
  storeLDS();
  for (int kk = 0; kk < FEAT; kk += BK) {
    __syncthreads();                        // staged tile visible to all waves
    if (kk + BK < FEAT) loadRegs(kk + BK);  // prefetch overlaps the WMMAs below
#pragma unroll
    for (int kc = 0; kc < 2; ++kc) {
      v16h a[2], b[2];
#pragma unroll
      for (int i = 0; i < 2; ++i)
        a[i] = load_frag16(&As[(wm + i * 16 + r) * LDA + kc * 32 + h * 8]);
#pragma unroll
      for (int j = 0; j < 2; ++j)
        b[j] = load_frag16(&Bs[(wn + j * 16 + r) * LDA + kc * 32 + h * 8]);
#pragma unroll
      for (int i = 0; i < 2; ++i)
#pragma unroll
        for (int j = 0; j < 2; ++j)
          c[i][j] = __builtin_amdgcn_wmma_f32_16x16x32_f16(
              false, a[i], false, b[j], (short)0, c[i][j], false, false);
    }
    __syncthreads();                        // everyone done reading before overwrite
    if (kk + BK < FEAT) storeLDS();
  }

  if constexpr (!TRANS_OUT) {
#pragma unroll
    for (int i = 0; i < 2; ++i)
#pragma unroll
      for (int j = 0; j < 2; ++j) {
        int gn = n0 + wn + j * 16 + r;
        float bj = HAS_BIAS ? bias[gn] : 0.f;
#pragma unroll
        for (int v = 0; v < 8; ++v) {
          int gm = m0 + wm + i * 16 + v + h * 8;
          Out[(long long)gm * FEAT + gn] = (half_t)(c[i][j][v] + bj);
        }
      }
  } else {
    // LDS transpose so VpT stores are coalesced 16B chunks.
    __shared__ half_t Cs[BN * (BM + 8)];
    constexpr int LDC = BM + 8;  // 136
#pragma unroll
    for (int i = 0; i < 2; ++i)
#pragma unroll
      for (int j = 0; j < 2; ++j) {
        int ln = wn + j * 16 + r;
#pragma unroll
        for (int v = 0; v < 8; ++v) {
          int lm = wm + i * 16 + v + h * 8;
          Cs[ln * LDC + lm] = (half_t)c[i][j][v];
        }
      }
    __syncthreads();
#pragma unroll
    for (int it = 0; it < 4; ++it) {
      int id = tid + it * 256;            // 0..1023
      int row = id >> 4;                  // 0..63 : local n (out row j)
      int col = (id & 15) * 8;            // 0..120: local m
      v8h vv = *(const v8h*)&Cs[row * LDC + col];
      *(v8h*)(Out + (long long)(n0 + row) * NN + m0 + col) = vv;
    }
  }
}

// ---------------------------------------------------------------------------
// Kernel 3: fused attention. Block = (16 query rows, 1 group), 8 waves.
//  S[16][1024] fp32 lives entirely in LDS (320KB WGP LDS) -> two-pass softmax.
//  K / VpT tiles staged via async global->LDS copies, double buffered so the
//  next stage's HBM traffic overlaps the current stage's WMMAs.
//  Phase1: S = (Q_g K_g^T)*rsqrt(128) + logw
//  Phase2: row softmax, P stored f16 in LDS
//  Phase3: O = P @ VpT_g^T, + conv_b -> out fp32
// ---------------------------------------------------------------------------
__global__ __launch_bounds__(256) void attn_kernel(
    const half_t* __restrict__ Q, const half_t* __restrict__ Kt,
    const half_t* __restrict__ VpT, const float* __restrict__ logw,
    const float* __restrict__ conv_b, float* __restrict__ out) {
  constexpr int ROWS = 16;
  constexpr int LDS_S = 1028;  // f32 row stride (pad 4 -> conflict-free)
  constexpr int LDS_P = 1032;  // f16 row stride (pad 8 -> 16B aligned, conflict-free)
  constexpr int LDT   = 136;   // f16 tile row stride
  __shared__ float  S[ROWS * LDS_S];   //  65792 B
  __shared__ half_t P[ROWS * LDS_P];   //  33024 B
  __shared__ half_t T[2][128 * LDT];   //  69632 B  (~164.5 KB total)

  int tid = threadIdx.x, wave = tid >> 5, lane = tid & 31;
  int r = lane & 15, h = lane >> 4;
  int n0 = blockIdx.x * ROWS;
  int g  = blockIdx.y;
  const float scale = 0.088388347648318447f;  // 1/sqrt(128)

  // Q fragments for rows n0..n0+15, features g*128..+127 (4 K-chunks), in VGPRs.
  v16h qa[4];
#pragma unroll
  for (int kc = 0; kc < 4; ++kc)
    qa[kc] = load_frag16(Q + (long long)(n0 + r) * FEAT + g * DIMG + kc * 32 + h * 8);

  auto issueK = [&](int s, int buf) {  // stage K rows m=s*128..+127, cols g*128..+127
#pragma unroll
    for (int it = 0; it < 8; ++it) {
      int id = tid + it * 256;          // 0..2047
      int ml = id >> 4, kcol = (id & 15) * 8;
      stage16(&T[buf][ml * LDT + kcol],
              Kt + (long long)(s * 128 + ml) * FEAT + g * DIMG + kcol);
    }
  };
  auto issueV = [&](int s, int buf) {  // stage VpT rows d=0..127, cols m=s*128..+127
#pragma unroll
    for (int it = 0; it < 8; ++it) {
      int id = tid + it * 256;
      int d = id >> 4, mcol = (id & 15) * 8;
      stage16(&T[buf][d * LDT + mcol],
              VpT + (long long)(g * DIMG + d) * NN + s * 128 + mcol);
    }
  };

  // ---- Phase 1: scores ----
  issueK(0, 0);
  for (int s = 0; s < 8; ++s) {
    wait_async_lds();                  // batch s (the only outstanding) complete
    __syncthreads();                   // visible to all; prev compute finished
    if (s < 7) issueK(s + 1, (s + 1) & 1);  // overlaps WMMAs below

    const half_t* Tb = T[s & 1];
    v8f acc = {};
#pragma unroll
    for (int kc = 0; kc < 4; ++kc) {
      v16h b = load_frag16(&Tb[(wave * 16 + r) * LDT + kc * 32 + h * 8]);
      acc = __builtin_amdgcn_wmma_f32_16x16x32_f16(
          false, qa[kc], false, b, (short)0, acc, false, false);
    }
    int mg = s * 128 + wave * 16 + r;  // key index for this lane
#pragma unroll
    for (int v = 0; v < 8; ++v) {
      int rr = v + h * 8;
      float lw = logw[((long long)(n0 + rr) * GROUP + g) * 1024 + mg];
      S[rr * LDS_S + mg] = acc[v] * scale + lw;
    }
  }
  __syncthreads();

  // ---- Phase 2: softmax (wave w handles rows 2w, 2w+1) ----
#pragma unroll
  for (int rw = 0; rw < 2; ++rw) {
    int row = wave * 2 + rw;
    float* Sr = &S[row * LDS_S];
    half_t* Pr = &P[row * LDS_P];
    float mx = -1e30f;
    for (int i = lane; i < 1024; i += 32) mx = fmaxf(mx, Sr[i]);
#pragma unroll
    for (int off = 16; off > 0; off >>= 1) mx = fmaxf(mx, __shfl_xor(mx, off, 32));
    float sum = 0.f;
    for (int i = lane; i < 1024; i += 32) {
      float e = __expf(Sr[i] - mx);
      Pr[i] = (half_t)e;
      sum += e;
    }
#pragma unroll
    for (int off = 16; off > 0; off >>= 1) sum += __shfl_xor(sum, off, 32);
    float inv = 1.f / sum;
    for (int i = lane; i < 1024; i += 32) Pr[i] = (half_t)((float)Pr[i] * inv);
  }
  __syncthreads();

  // ---- Phase 3: O = P @ Vp_g (wave w owns output cols d = 16w..16w+15) ----
  v8f oacc = {};
  issueV(0, 0);
  for (int s = 0; s < 8; ++s) {
    wait_async_lds();
    __syncthreads();
    if (s < 7) issueV(s + 1, (s + 1) & 1);

    const half_t* Tb = T[s & 1];
#pragma unroll
    for (int kc = 0; kc < 4; ++kc) {
      v16h a = load_frag16(&P[r * LDS_P + s * 128 + kc * 32 + h * 8]);
      v16h b = load_frag16(&Tb[(wave * 16 + r) * LDT + kc * 32 + h * 8]);
      oacc = __builtin_amdgcn_wmma_f32_16x16x32_f16(
          false, a, false, b, (short)0, oacc, false, false);
    }
  }

  int dg = g * DIMG + wave * 16 + r;
  float bj = conv_b[dg];
#pragma unroll
  for (int v = 0; v < 8; ++v) {
    int rr = v + h * 8;
    out[(long long)(n0 + rr) * FEAT + dg] = oacc[v] + bj;
  }
}

// ---------------------------------------------------------------------------
extern "C" void kernel_launch(void* const* d_in, const int* in_sizes, int n_in,
                              void* d_out, int out_size, void* d_ws, size_t ws_size,
                              hipStream_t stream) {
  const float* roi    = (const float*)d_in[0];
  const float* pe     = (const float*)d_in[1];
  const float* fc_w   = (const float*)d_in[2];
  const float* fc_b   = (const float*)d_in[3];
  const float* fc2_w  = (const float*)d_in[4];
  const float* fc2_b  = (const float*)d_in[5];
  const float* fc3_w  = (const float*)d_in[6];
  const float* fc3_b  = (const float*)d_in[7];
  const float* conv_w = (const float*)d_in[8];
  const float* conv_b = (const float*)d_in[9];
  float* out = (float*)d_out;

  char* ws = (char*)d_ws;
  half_t* Q16  = (half_t*)(ws);                               // 4 MB
  half_t* K16  = (half_t*)(ws + (size_t)NN * FEAT * 2);       // 4 MB
  half_t* VpT  = (half_t*)(ws + (size_t)NN * FEAT * 4);       // 4 MB (2048 x 1024 f16)
  float*  logw = (float*) (ws + (size_t)NN * FEAT * 6);       // 64 MB

  posbias_kernel<<<dim3(4096), dim3(256), 0, stream>>>(pe, fc_w, fc_b, logw);
  gemm_nt_kernel<false, true ><<<dim3(8, 32), dim3(256), 0, stream>>>(roi, fc2_w, fc2_b, Q16);
  gemm_nt_kernel<false, true ><<<dim3(8, 32), dim3(256), 0, stream>>>(roi, fc3_w, fc3_b, K16);
  gemm_nt_kernel<true,  false><<<dim3(8, 32), dim3(256), 0, stream>>>(roi, conv_w, nullptr, VpT);
  attn_kernel<<<dim3(64, 16), dim3(256), 0, stream>>>(Q16, K16, VpT, logw, conv_b, out);
}
